// EncoderBlock_32392643346964
// MI455X (gfx1250) — matrix operand order repaired and checked by reading, not compiled, not run on hardware
//
#include <hip/hip_runtime.h>

// ---------------------------------------------------------------------------
// Types for CDNA5 WMMA (gfx1250, wave32): D = A(16x32 bf16) x B(32x16 bf16) + C(f32)
// ---------------------------------------------------------------------------
typedef __attribute__((ext_vector_type(16))) __bf16 v16bf;
typedef __attribute__((ext_vector_type(8)))  float  v8f;
typedef int v4i __attribute__((vector_size(16)));

union FragAB { v16bf v; uint4 q[2]; };

#define AS1 __attribute__((address_space(1)))
#define AS3 __attribute__((address_space(3)))

#if __has_builtin(__builtin_amdgcn_global_load_async_to_lds_b128)
#define HAVE_ASYNC_LDS 1
// builtin signature (probe-verified): (v4i AS1* gsrc, v4i AS3* ldst, Ii off, Ii cpol)
#define GLDS_B128(gsrc, ldst) \
    __builtin_amdgcn_global_load_async_to_lds_b128((v4i AS1*)(gsrc), (v4i AS3*)(ldst), 0, 0)
#else
#define HAVE_ASYNC_LDS 0
#endif

__device__ __forceinline__ void wait_async_lds() {
#if __has_builtin(__builtin_amdgcn_s_wait_asynccnt)
    __builtin_amdgcn_s_wait_asynccnt(0);
#else
    asm volatile("s_wait_asynccnt 0x0" ::: "memory");
#endif
}

__device__ __forceinline__ unsigned short f2bf(float f) {
    unsigned int u = __float_as_uint(f);
    u += 0x7FFFu + ((u >> 16) & 1u);      // round-to-nearest-even
    return (unsigned short)(u >> 16);
}

// ---------------------------------------------------------------------------
// RMSNorm over D, writes bf16 (for WMMA A operand) and optional f32 copy.
// ---------------------------------------------------------------------------
__global__ __launch_bounds__(256) void k_rmsnorm(
    const float* __restrict__ in, const float* __restrict__ w,
    unsigned short* __restrict__ outb, float* __restrict__ outf, int D)
{
    __shared__ float red[256];
    const size_t base = (size_t)blockIdx.x * D;
    float ss = 0.f;
    for (int i = threadIdx.x; i < D; i += 256) { float v = in[base + i]; ss += v * v; }
    red[threadIdx.x] = ss; __syncthreads();
    for (int o = 128; o > 0; o >>= 1) {
        if (threadIdx.x < o) red[threadIdx.x] += red[threadIdx.x + o];
        __syncthreads();
    }
    const float r = rsqrtf(red[0] / (float)D + 1e-6f);
    for (int i = threadIdx.x; i < D; i += 256) {
        float v = in[base + i] * r * w[i];
        outb[base + i] = f2bf(v);
        if (outf) outf[base + i] = v;
    }
}

// ---------------------------------------------------------------------------
// f32 [R,C] -> bf16 [C,R]  (weights into the [N,K] layout WMMA B-frags want)
// ---------------------------------------------------------------------------
__global__ __launch_bounds__(256) void k_castT(
    const float* __restrict__ in, unsigned short* __restrict__ out,
    int R, int C, size_t inB, size_t outB)
{
    __shared__ unsigned short tile[32][33];
    const float* ib = in + (size_t)blockIdx.z * inB;
    unsigned short* ob = out + (size_t)blockIdx.z * outB;
    const int c0 = blockIdx.x * 32, r0 = blockIdx.y * 32;
    const int tx = threadIdx.x & 31, ty = threadIdx.x >> 5;
    #pragma unroll
    for (int i = 0; i < 32; i += 8)
        tile[ty + i][tx] = f2bf(ib[(size_t)(r0 + ty + i) * C + (c0 + tx)]);
    __syncthreads();
    #pragma unroll
    for (int i = 0; i < 32; i += 8)
        ob[(size_t)(c0 + ty + i) * R + (r0 + tx)] = tile[tx][ty + i];
}

// ---------------------------------------------------------------------------
// Generic WMMA GEMM, LDS-staged + double buffered via async global->LDS.
// C[M,N](+epilogue) = A[M,K](bf16,row-major) x Bt[N,K](bf16,row-major)
// block = 256 thr = 8 waves; block tile 64x128; wave tile 16x64; K-step 32,
// explicitly software-pipelined two steps per iteration (K multiple of 64)
// so buffer addresses are compile-time LDS offsets.
// EPI 0: C = acc                        (z-batched)
// EPI 1: Out[oRidx[m]] += oScale[m]*acc (MoE scatter-accumulate, skip w==0)
// EPI 2: Out[m] = Res[m] + gamma[n]*acc (attention layerscale residual)
// ---------------------------------------------------------------------------
template<int EPI>
__global__ __launch_bounds__(256) void k_gemm(
    const unsigned short* __restrict__ A, size_t lda, size_t batchA,
    const int* __restrict__ aRidx,
    const unsigned short* __restrict__ Bt, size_t ldb, size_t batchB,
    float* __restrict__ C, size_t ldc, size_t batchC,
    float* __restrict__ Out, size_t ldo,
    const int* __restrict__ oRidx, const float* __restrict__ oScale,
    const float* __restrict__ Res, const float* __restrict__ gamma,
    const int* __restrict__ mCount, int M, int N, int K)
{
    constexpr int APAD = 40;               // padded row stride (el): 80B walks 20 banks
    constexpr int BPAD = 40;
    constexpr int ATILE = 64 * APAD;       // 2560 el
    constexpr int BTILE = 128 * BPAD;      // 5120 el
    constexpr int BUFSZ = ATILE + BTILE;   // 7680 el (15360 B); x2 buffers = 30 KB
    __shared__ unsigned short smem[2 * BUFSZ];

    const int m0 = blockIdx.y * 64;
    if (mCount) { const int lim = (*mCount + 63) & ~63; if (m0 >= lim) return; }
    const int n0   = blockIdx.x * 128;
    const int z    = blockIdx.z;
    const int tid  = threadIdx.x;
    const int lane = tid & 31;
    const int wid  = tid >> 5;
    const int wm   = (wid & 3) << 4;   // 4 waves stacked in M
    const int wn   = (wid >> 2) << 6;  // 2 wave groups in N
    const int hi   = lane >> 4;
    const int l15  = lane & 15;

    const unsigned short* Ab = A + (size_t)z * batchA;
    const unsigned short* Bb = Bt + (size_t)z * batchB;

    // --- cooperative copy assignments: B = 128 rows x 2 x 16B (all 256 thr),
    //                                   A =  64 rows x 2 x 16B (thr 0..127)
    const int bn = tid >> 1, bc = tid & 1;
    int nB_ = n0 + bn; if (nB_ >= N) nB_ = N - 1;          // clamp keeps loads legal
    const unsigned short* bSrc = Bb + (size_t)nB_ * ldb + bc * 8;
    const int bDst = bn * BPAD + bc * 8;
    const unsigned short* aSrc = nullptr;
    const int aDst = (tid >> 1) * APAD + (tid & 1) * 8;
    if (tid < 128) {
        const int am2 = m0 + (tid >> 1);
        const size_t arow2 = aRidx ? (size_t)aRidx[am2] : (size_t)am2;
        aSrc = Ab + arow2 * lda + (tid & 1) * 8;
    }

    unsigned short* const bufA0 = smem;
    unsigned short* const bufB0 = smem + ATILE;
    unsigned short* const bufA1 = smem + BUFSZ;
    unsigned short* const bufB1 = smem + BUFSZ + ATILE;

    v8f acc[4];
    #pragma unroll
    for (int j = 0; j < 4; ++j)
        #pragma unroll
        for (int r = 0; r < 8; ++r) acc[j][r] = 0.0f;

    // per-wave fragment base offsets within a buffer (compile-time strides)
    const int aOff = (wm + l15) * APAD + hi * 16;
    const int bOff = (wn + l15) * BPAD + hi * 16;

    // one K-step: load A frag + all 4 B frags into distinct regs, then 4 WMMAs
    auto mm_step = [&](const unsigned short* cA, const unsigned short* cB) {
        FragAB a, b0, b1, b2, b3;
        const unsigned short* ap = cA + aOff;
        a.q[0]  = *(const uint4*)(ap);
        a.q[1]  = *(const uint4*)(ap + 8);
        const unsigned short* bp = cB + bOff;
        b0.q[0] = *(const uint4*)(bp);
        b0.q[1] = *(const uint4*)(bp + 8);
        b1.q[0] = *(const uint4*)(bp + 16 * BPAD);
        b1.q[1] = *(const uint4*)(bp + 16 * BPAD + 8);
        b2.q[0] = *(const uint4*)(bp + 32 * BPAD);
        b2.q[1] = *(const uint4*)(bp + 32 * BPAD + 8);
        b3.q[0] = *(const uint4*)(bp + 48 * BPAD);
        b3.q[1] = *(const uint4*)(bp + 48 * BPAD + 8);
        acc[0] = __builtin_amdgcn_wmma_f32_16x16x32_bf16(false, a.v, false, b0.v, (short)0, acc[0], false, false);
        acc[1] = __builtin_amdgcn_wmma_f32_16x16x32_bf16(false, a.v, false, b1.v, (short)0, acc[1], false, false);
        acc[2] = __builtin_amdgcn_wmma_f32_16x16x32_bf16(false, a.v, false, b2.v, (short)0, acc[2], false, false);
        acc[3] = __builtin_amdgcn_wmma_f32_16x16x32_bf16(false, a.v, false, b3.v, (short)0, acc[3], false, false);
    };

    auto stage = [&](int k, unsigned short* dA, unsigned short* dB) {
#if HAVE_ASYNC_LDS
        GLDS_B128(bSrc + k, dB + bDst);
        if (tid < 128) GLDS_B128(aSrc + k, dA + aDst);
#else
        uint4 rB = *(const uint4*)(bSrc + k);
        *(uint4*)(dB + bDst) = rB;
        if (tid < 128) {
            uint4 rA = *(const uint4*)(aSrc + k);
            *(uint4*)(dA + aDst) = rA;
        }
#endif
    };

    // --- prologue: stage K-tile 0 into buffer 0
    stage(0, bufA0, bufB0);
#if HAVE_ASYNC_LDS
    wait_async_lds();
#endif
    __syncthreads();

    // --- main loop: two K-steps per iteration, fixed buffer roles (K % 64 == 0)
    for (int k0 = 0; k0 < K; k0 += 64) {
        // step A: read buf0 (tile k0); stage tile k0+32 -> buf1 (always in range)
        stage(k0 + 32, bufA1, bufB1);
        mm_step(bufA0, bufB0);
#if HAVE_ASYNC_LDS
        wait_async_lds();
#endif
        __syncthreads();

        // step B: read buf1 (tile k0+32); stage tile k0+64 -> buf0 if in range
        const bool more = (k0 + 64) < K;
        if (more) stage(k0 + 64, bufA0, bufB0);
        mm_step(bufA1, bufB1);
        if (more) {
#if HAVE_ASYNC_LDS
            wait_async_lds();
#endif
            __syncthreads();
        }
    }

    // C frag layout: VGPR r -> M = r + 8*hi ; N = lane&15
    #pragma unroll
    for (int j = 0; j < 4; ++j) {
        const int n = n0 + wn + j * 16 + l15;
        if (n >= N) continue;
        #pragma unroll
        for (int r = 0; r < 8; ++r) {
            const int m = m0 + wm + r + hi * 8;
            const float v = acc[j][r];
            if (EPI == 0) {
                C[(size_t)z * batchC + (size_t)m * ldc + n] = v;
            } else if (EPI == 1) {
                const float s = oScale[m];
                if (s != 0.0f) {
                    float* p = Out + (size_t)oRidx[m] * ldo + n;
                    *p += s * v;
                }
            } else {
                Out[(size_t)m * ldo + n] = Res[(size_t)m * ldo + n] + gamma[n] * v;
            }
        }
    }
}

// ---------------------------------------------------------------------------
// Per-head RMSNorm of q,k + bf16 cast; v cast + transpose to [Z, HD, S].
// ---------------------------------------------------------------------------
__global__ __launch_bounds__(64) void k_qkv_post(
    const float* __restrict__ qf, const float* __restrict__ kf, const float* __restrict__ vf,
    const float* __restrict__ qw, const float* __restrict__ kw,
    unsigned short* __restrict__ qb, unsigned short* __restrict__ kb,
    unsigned short* __restrict__ vtb, int S, int H)
{
    __shared__ float red[64];
    const int t = blockIdx.x, h = blockIdx.y, d = threadIdx.x;
    const int b = t / S, s = t % S;
    const int zh = b * H + h;
    const size_t iin = (size_t)t * (H * 64) + h * 64 + d;

    const float qv = qf[iin];
    red[d] = qv * qv; __syncthreads();
    for (int o = 32; o > 0; o >>= 1) { if (d < o) red[d] += red[d + o]; __syncthreads(); }
    const float qr = rsqrtf(red[0] / 64.0f + 1e-6f);
    qb[((size_t)zh * S + s) * 64 + d] = f2bf(qv * qr * qw[d]);
    __syncthreads();

    const float kv = kf[iin];
    red[d] = kv * kv; __syncthreads();
    for (int o = 32; o > 0; o >>= 1) { if (d < o) red[d] += red[d + o]; __syncthreads(); }
    const float kr = rsqrtf(red[0] / 64.0f + 1e-6f);
    kb[((size_t)zh * S + s) * 64 + d] = f2bf(kv * kr * kw[d]);

    vtb[((size_t)zh * 64 + d) * S + s] = f2bf(vf[iin]);
}

// ---------------------------------------------------------------------------
// Row softmax of scores (pre-scaled) -> bf16 probs.  one block per row.
// ---------------------------------------------------------------------------
__global__ __launch_bounds__(256) void k_softmax(
    const float* __restrict__ sc, unsigned short* __restrict__ pb, int S, float scale)
{
    __shared__ float red[256];
    const size_t base = (size_t)blockIdx.x * S;
    const int tid = threadIdx.x;
    float v[4];
    float mx = -3.0e38f;
    #pragma unroll
    for (int i = 0; i < 4; ++i) { v[i] = sc[base + tid + i * 256] * scale; mx = fmaxf(mx, v[i]); }
    red[tid] = mx; __syncthreads();
    for (int o = 128; o > 0; o >>= 1) { if (tid < o) red[tid] = fmaxf(red[tid], red[tid + o]); __syncthreads(); }
    mx = red[0]; __syncthreads();
    float sum = 0.f;
    #pragma unroll
    for (int i = 0; i < 4; ++i) { v[i] = __expf(v[i] - mx); sum += v[i]; }
    red[tid] = sum; __syncthreads();
    for (int o = 128; o > 0; o >>= 1) { if (tid < o) red[tid] += red[tid + o]; __syncthreads(); }
    const float inv = 1.0f / red[0];
    #pragma unroll
    for (int i = 0; i < 4; ++i) pb[base + tid + i * 256] = f2bf(v[i] * inv);
}

// ---------------------------------------------------------------------------
// attn [Z=B*H, S, HD] f32 -> ob bf16 [B*S, H*HD]   (rows for the Wo GEMM)
// ---------------------------------------------------------------------------
__global__ __launch_bounds__(256) void k_rows_cast(
    const float* __restrict__ attn, unsigned short* __restrict__ ob,
    int Bc, int Hc, int Sc, int HDc, size_t total)
{
    size_t g = (size_t)blockIdx.x * 256 + threadIdx.x;
    if (g >= total) return;
    const int d = (int)(g % HDc); size_t r = g / HDc;
    const int s = (int)(r % Sc);  r /= Sc;
    const int h = (int)(r % Hc);
    const int b = (int)(r / Hc);
    ob[((size_t)(b * Sc + s)) * (Hc * HDc) + h * HDc + d] = f2bf(attn[g]);
}

// ---------------------------------------------------------------------------
// Router: logits = xt @ gate (D x 8); raw logits -> d_out; top-2 softmax
// weights; build per-expert token buckets via atomics.  one block per token.
// ---------------------------------------------------------------------------
__global__ __launch_bounds__(256) void k_router(
    const float* __restrict__ xt, const float* __restrict__ gate,
    float* __restrict__ logits, int* __restrict__ counts,
    int* __restrict__ bIdx, float* __restrict__ bScale, int T, int D)
{
    __shared__ float red[256 * 8];
    const int t = blockIdx.x;
    float p[8];
    #pragma unroll
    for (int e = 0; e < 8; ++e) p[e] = 0.f;
    for (int i = threadIdx.x; i < D; i += 256) {
        const float x = xt[(size_t)t * D + i];
        const float* g = gate + (size_t)i * 8;
        #pragma unroll
        for (int e = 0; e < 8; ++e) p[e] += x * g[e];
    }
    #pragma unroll
    for (int e = 0; e < 8; ++e) red[threadIdx.x * 8 + e] = p[e];
    __syncthreads();
    for (int o = 128; o > 0; o >>= 1) {
        if (threadIdx.x < o)
            #pragma unroll
            for (int e = 0; e < 8; ++e) red[threadIdx.x * 8 + e] += red[(threadIdx.x + o) * 8 + e];
        __syncthreads();
    }
    if (threadIdx.x < 8) logits[(size_t)t * 8 + threadIdx.x] = red[threadIdx.x];
    if (threadIdx.x == 0) {
        float l[8], mx = -3.0e38f;
        #pragma unroll
        for (int e = 0; e < 8; ++e) { l[e] = red[e]; mx = fmaxf(mx, l[e]); }
        float ex[8];
        #pragma unroll
        for (int e = 0; e < 8; ++e) ex[e] = __expf(l[e] - mx);
        int i0 = 0;
        #pragma unroll
        for (int e = 1; e < 8; ++e) if (l[e] > l[i0]) i0 = e;
        int i1 = (i0 == 0) ? 1 : 0;
        #pragma unroll
        for (int e = 0; e < 8; ++e) if (e != i0 && l[e] > l[i1]) i1 = e;
        const float ws = ex[i0] + ex[i1];
        const float w0 = ex[i0] / ws, w1 = ex[i1] / ws;
        int p0 = atomicAdd(&counts[i0], 1);
        bIdx[i0 * T + p0] = t; bScale[i0 * T + p0] = w0;
        int p1 = atomicAdd(&counts[i1], 1);
        bIdx[i1 * T + p1] = t; bScale[i1 * T + p1] = w1;
    }
}

// ---------------------------------------------------------------------------
// h = silu(g) * u -> bf16, only for rows < count (bucket-position indexed)
// ---------------------------------------------------------------------------
__global__ __launch_bounds__(256) void k_silu_mul(
    const float* __restrict__ g, const float* __restrict__ u,
    unsigned short* __restrict__ hb, const int* __restrict__ count, int F)
{
    const int m = blockIdx.y;
    if (m >= *count) return;
    const int n = blockIdx.x * 256 + threadIdx.x;
    const size_t i = (size_t)m * F + n;
    const float gv = g[i];
    hb[i] = f2bf(gv / (1.0f + __expf(-gv)) * u[i]);
}

// ---------------------------------------------------------------------------
// Launch
// ---------------------------------------------------------------------------
static inline dim3 gemm_grid(int M, int N, int Z) {
    return dim3((unsigned)((N + 127) / 128), (unsigned)((M + 63) / 64), (unsigned)Z);
}

extern "C" void kernel_launch(void* const* d_in, const int* in_sizes, int n_in,
                              void* d_out, int out_size, void* d_ws, size_t ws_size,
                              hipStream_t stream)
{
    (void)in_sizes; (void)n_in; (void)out_size; (void)ws_size;
    constexpr int B = 4, S = 1024, D = 1024, H = 16, HD = 64, F = 4096, E = 8;
    constexpr int T = B * S, Z = B * H;

    const float* hidden = (const float*)d_in[0];
    const float* ln1_w  = (const float*)d_in[1];
    const float* ln2_w  = (const float*)d_in[2];
    const float* Wq     = (const float*)d_in[3];
    const float* Wk     = (const float*)d_in[4];
    const float* Wv     = (const float*)d_in[5];
    const float* Wo     = (const float*)d_in[6];
    const float* qn_w   = (const float*)d_in[7];
    const float* kn_w   = (const float*)d_in[8];
    const float* gamma  = (const float*)d_in[9];
    const float* gate_w = (const float*)d_in[10];
    const float* Wg     = (const float*)d_in[11];
    const float* Wu     = (const float*)d_in[12];
    const float* Wd     = (const float*)d_in[13];

    float* outHidden = (float*)d_out;             // [T, D]
    float* outLogits = outHidden + (size_t)T * D; // [T, E]

    // ---- workspace carve-up (256B aligned) ----
    size_t off = 0;
    auto alloc = [&](size_t bytes) -> char* {
        off = (off + 255) & ~(size_t)255;
        char* p = (char*)d_ws + off;
        off += bytes;
        return p;
    };
    unsigned short* WqT = (unsigned short*)alloc((size_t)D * D * 2);
    unsigned short* WkT = (unsigned short*)alloc((size_t)D * D * 2);
    unsigned short* WvT = (unsigned short*)alloc((size_t)D * D * 2);
    unsigned short* WoT = (unsigned short*)alloc((size_t)D * D * 2);
    unsigned short* WgT = (unsigned short*)alloc((size_t)E * D * F * 2); // [e][F,D]
    unsigned short* WuT = (unsigned short*)alloc((size_t)E * D * F * 2); // [e][F,D]
    unsigned short* WdT = (unsigned short*)alloc((size_t)E * F * D * 2); // [e][D,F]
    unsigned short* xb  = (unsigned short*)alloc((size_t)T * D * 2);
    float* qf  = (float*)alloc((size_t)T * D * 4);
    float* kf  = (float*)alloc((size_t)T * D * 4);
    float* vf  = (float*)alloc((size_t)T * D * 4);
    unsigned short* qb  = (unsigned short*)alloc((size_t)Z * S * HD * 2);
    unsigned short* kb  = (unsigned short*)alloc((size_t)Z * S * HD * 2);
    unsigned short* vtb = (unsigned short*)alloc((size_t)Z * HD * S * 2);
    float* scores = (float*)alloc((size_t)Z * S * S * 4);                  // 256 MB
    unsigned short* probs = (unsigned short*)alloc((size_t)Z * S * S * 2); // 128 MB
    float* attn = (float*)alloc((size_t)Z * S * HD * 4);
    unsigned short* ob  = (unsigned short*)alloc((size_t)T * D * 2);
    float* xt2 = (float*)alloc((size_t)T * D * 4);
    unsigned short* xb2 = (unsigned short*)alloc((size_t)T * D * 2);
    int*   counts = (int*)alloc((size_t)E * 4);
    int*   bIdx   = (int*)alloc((size_t)E * T * 4);
    float* bScale = (float*)alloc((size_t)E * T * 4);
    // MoE intermediates alias the (dead by then) scores/probs regions
    float* gbuf = scores;                                   // [T,F] f32 (64 MB)
    float* ubuf = scores + (size_t)T * F;                   // [T,F] f32 (64 MB)
    unsigned short* hbuf = probs;                           // [T,F] bf16 (32 MB)

    // ---- weight cast+transpose to bf16 [N,K] ----
    k_castT<<<dim3(32, 32, 1), 256, 0, stream>>>(Wq, WqT, D, D, 0, 0);
    k_castT<<<dim3(32, 32, 1), 256, 0, stream>>>(Wk, WkT, D, D, 0, 0);
    k_castT<<<dim3(32, 32, 1), 256, 0, stream>>>(Wv, WvT, D, D, 0, 0);
    k_castT<<<dim3(32, 32, 1), 256, 0, stream>>>(Wo, WoT, D, D, 0, 0);
    k_castT<<<dim3(F / 32, D / 32, E), 256, 0, stream>>>(Wg, WgT, D, F, (size_t)D * F, (size_t)D * F);
    k_castT<<<dim3(F / 32, D / 32, E), 256, 0, stream>>>(Wu, WuT, D, F, (size_t)D * F, (size_t)D * F);
    k_castT<<<dim3(D / 32, F / 32, E), 256, 0, stream>>>(Wd, WdT, F, D, (size_t)F * D, (size_t)F * D);

    // ---- attention branch ----
    k_rmsnorm<<<T, 256, 0, stream>>>(hidden, ln1_w, xb, nullptr, D);

    k_gemm<0><<<gemm_grid(T, D, 1), 256, 0, stream>>>(
        xb, D, 0, nullptr, WqT, D, 0, qf, D, 0,
        nullptr, 0, nullptr, nullptr, nullptr, nullptr, nullptr, T, D, D);
    k_gemm<0><<<gemm_grid(T, D, 1), 256, 0, stream>>>(
        xb, D, 0, nullptr, WkT, D, 0, kf, D, 0,
        nullptr, 0, nullptr, nullptr, nullptr, nullptr, nullptr, T, D, D);
    k_gemm<0><<<gemm_grid(T, D, 1), 256, 0, stream>>>(
        xb, D, 0, nullptr, WvT, D, 0, vf, D, 0,
        nullptr, 0, nullptr, nullptr, nullptr, nullptr, nullptr, T, D, D);

    k_qkv_post<<<dim3(T, H), 64, 0, stream>>>(qf, kf, vf, qn_w, kn_w, qb, kb, vtb, S, H);

    // scores[z] = Q[z] (SxHD) x K[z]^T  — K stored [S,HD] is exactly Bt=[N,K]
    k_gemm<0><<<gemm_grid(S, S, Z), 256, 0, stream>>>(
        qb, HD, (size_t)S * HD, nullptr, kb, HD, (size_t)S * HD,
        scores, S, (size_t)S * S,
        nullptr, 0, nullptr, nullptr, nullptr, nullptr, nullptr, S, S, HD);

    k_softmax<<<Z * S, 256, 0, stream>>>(scores, probs, S, 0.125f /* HD^-0.5 */);

    // attn[z] = P[z] (SxS) x V[z]  — Bt = V^T stored [HD,S]
    k_gemm<0><<<gemm_grid(S, HD, Z), 256, 0, stream>>>(
        probs, S, (size_t)S * S, nullptr, vtb, S, (size_t)HD * S,
        attn, HD, (size_t)S * HD,
        nullptr, 0, nullptr, nullptr, nullptr, nullptr, nullptr, S, HD, S);

    k_rows_cast<<<(unsigned)(((size_t)Z * S * HD + 255) / 256), 256, 0, stream>>>(
        attn, ob, B, H, S, HD, (size_t)Z * S * HD);

    // hidden1 = residual + gamma_sa * (o @ Wo)  -> directly into d_out
    k_gemm<2><<<gemm_grid(T, D, 1), 256, 0, stream>>>(
        ob, D, 0, nullptr, WoT, D, 0, nullptr, 0, 0,
        outHidden, D, nullptr, nullptr, hidden, gamma, nullptr, T, D, D);

    // ---- MoE branch ----
    k_rmsnorm<<<T, 256, 0, stream>>>(outHidden, ln2_w, xb2, xt2, D);

    (void)hipMemsetAsync(counts, 0, (size_t)E * 4, stream);
    (void)hipMemsetAsync(bIdx,   0, (size_t)E * T * 4, stream);
    (void)hipMemsetAsync(bScale, 0, (size_t)E * T * 4, stream);

    k_router<<<T, 256, 0, stream>>>(xt2, gate_w, outLogits, counts, bIdx, bScale, T, D);

    for (int e = 0; e < E; ++e) {
        const unsigned short* WgTe = WgT + (size_t)e * D * F;
        const unsigned short* WuTe = WuT + (size_t)e * D * F;
        const unsigned short* WdTe = WdT + (size_t)e * F * D;
        const int* idx = bIdx + (size_t)e * T;
        const float* scl = bScale + (size_t)e * T;
        const int* cnt = counts + e;

        // g = gather(x) @ Wg[e] ; u = gather(x) @ Wu[e]    (rows = bucket positions)
        k_gemm<0><<<gemm_grid(T, F, 1), 256, 0, stream>>>(
            xb2, D, 0, idx, WgTe, D, 0, gbuf, F, 0,
            nullptr, 0, nullptr, nullptr, nullptr, nullptr, cnt, T, F, D);
        k_gemm<0><<<gemm_grid(T, F, 1), 256, 0, stream>>>(
            xb2, D, 0, idx, WuTe, D, 0, ubuf, F, 0,
            nullptr, 0, nullptr, nullptr, nullptr, nullptr, cnt, T, F, D);

        k_silu_mul<<<dim3(F / 256, T), 256, 0, stream>>>(gbuf, ubuf, hbuf, cnt, F);

        // out[token] += w * (h @ Wd[e])
        k_gemm<1><<<gemm_grid(T, D, 1), 256, 0, stream>>>(
            hbuf, F, 0, nullptr, WdTe, F, 0, nullptr, 0, 0,
            outHidden, D, idx, scl, nullptr, nullptr, cnt, T, D, F);
    }
}
